// VulBertaGGCN_62955630625164
// MI455X (gfx1250) — matrix operand order repaired and checked by reading, not compiled, not run on hardware
//
#include <hip/hip_runtime.h>
#include <hip/hip_bf16.h>

// ---------------------------------------------------------------------------
// VulBERTa-GGCN forward for MI455X (gfx1250, wave32, WMMA).
// All dense math runs through v_wmma_f32_16x16x32_bf16 (f32 accumulate),
// with 2x2 register blocking per wave (4 WMMAs per A/B fragment fetch).
// ---------------------------------------------------------------------------

#define BATCH   128
#define NPER    205
#define NNODE   (BATCH * NPER)   // 26240, multiple of 32
#define HDIM    200
#define HP      208              // H padded to 16 (f32 row stride)
#define KP      224              // H padded to 32 (WMMA K granularity)
#define NTP     14               // output col tiles padded even (14*16 = 224)
#define G3      608              // 3*H padded to 16
#define SEQ     512
#define EMB     768
#define NEDGE   209920
#define NLAYER  6

typedef __attribute__((ext_vector_type(16))) __bf16 v16bf;
typedef __attribute__((ext_vector_type(8)))  float  v8f;

// --------------------------- WMMA fragment helpers -------------------------
// A fragment (16x32 bf16, row-major source with row stride lda, k-base kb):
//   lane L holds row M = L&15.  Elements 0..7  : K = kb + (L>>4)*8 + j
//                               Elements 8..15 : K = kb + 16 + (L>>4)*8 + (j-8)
__device__ __forceinline__ v16bf load_a_frag(const __bf16* __restrict__ A,
                                             int row, int lda, int kb, int lane) {
  const int hi = lane >> 4;
  const __bf16* p0 = A + (size_t)row * lda + kb + hi * 8;
  const __bf16* p1 = p0 + 16;
  v16bf a;
#pragma unroll
  for (int j = 0; j < 8; ++j) { a[j] = p0[j]; a[8 + j] = p1[j]; }
  return a;
}

// B fragments are pre-packed in fragment order: one contiguous v16bf per lane.
__device__ __forceinline__ v16bf load_b_frag(const __bf16* __restrict__ Bp,
                                             int kt, int nt, int NT, int lane) {
  const v16bf* p = (const v16bf*)(Bp + (((size_t)(kt * NT + nt)) * 32 + lane) * 16);
  return *p;
}

__device__ __forceinline__ v8f wmma_bf16(v16bf a, v16bf b, v8f c) {
  return __builtin_amdgcn_wmma_f32_16x16x32_bf16(false, a, false, b,
                                                 (short)0, c, false, false);
}

// --------------------------- weight packing --------------------------------
// Packs a logical K x N weight matrix into per-fragment order, bf16, with
// zero padding outside the valid region.
//  mode 0: ggc layer     W(k,n) = src[k*200+n]            (k<200,n<200)
//  mode 1: gru (W^T)     W(k,n) = src[n*200+k]            (k<200,n<600)
//  mode 2: text conv     r=k: kk=r/768, ci=r%768; W = src[(n*768+ci)*K + kk] (n<200)
//  mode 3: conv0         r=k: kk=r/224, ci=r%224; W = src[n*600+ci*3+kk]     (n<200,ci<200)
__global__ void pack_b_kernel(__bf16* __restrict__ dst, const float* __restrict__ src,
                              int KT, int NT, int mode, int Kw) {
  long long idx = blockIdx.x * (long long)blockDim.x + threadIdx.x;
  long long total = (long long)KT * NT * 512;
  if (idx >= total) return;
  int  j    = (int)(idx & 15);
  int  lane = (int)((idx >> 4) & 31);
  long long tile = idx >> 9;
  int  nt = (int)(tile % NT);
  int  kt = (int)(tile / NT);
  int  k  = kt * 32 + (lane >> 4) * 8 + (j & 7) + ((j >> 3) << 4);
  int  n  = nt * 16 + (lane & 15);
  float v = 0.0f;
  if (mode == 0)      { if (k < HDIM && n < HDIM) v = src[k * HDIM + n]; }
  else if (mode == 1) { if (k < HDIM && n < 3*HDIM) v = src[n * HDIM + k]; }
  else if (mode == 2) { int kk = k / EMB, ci = k % EMB;
                        if (n < HDIM && kk < Kw) v = src[((size_t)n * EMB + ci) * Kw + kk]; }
  else                { int kk = k / KP,  ci = k % KP;
                        if (n < HDIM && ci < HDIM && kk < 3) v = src[(size_t)n * 600 + ci * 3 + kk]; }
  dst[idx] = (__bf16)v;
}

// --------------------------- misc elementwise ------------------------------
__global__ void zero_f32_kernel(float* __restrict__ p, long long n) {
  long long i = blockIdx.x * (long long)blockDim.x + threadIdx.x;
  if (i < n) p[i] = 0.0f;
}

__global__ void init_h_kernel(const float* __restrict__ x,
                              float* __restrict__ Hf, __bf16* __restrict__ Hb) {
  long long i = blockIdx.x * (long long)blockDim.x + threadIdx.x;
  if (i >= (long long)NNODE * KP) return;
  int c = (int)(i % KP);
  long long row = i / KP;
  float v = (c < HDIM) ? x[row * HDIM + c] : 0.0f;
  Hb[i] = (__bf16)v;
  if (c < HP) Hf[row * HP + c] = v;
}

__global__ void cvt_agg_kernel(const float* __restrict__ agg, __bf16* __restrict__ aggb) {
  long long i = blockIdx.x * (long long)blockDim.x + threadIdx.x;
  if (i >= (long long)NNODE * KP) return;
  int c = (int)(i % KP);
  long long row = i / KP;
  float v = (c < HDIM) ? agg[row * HP + c] : 0.0f;
  aggb[i] = (__bf16)v;
}

__global__ void embed_gather_kernel(const int* __restrict__ ids,
                                    const float* __restrict__ tab,
                                    __bf16* __restrict__ e) {
  long long i = blockIdx.x * (long long)blockDim.x + threadIdx.x;
  if (i >= (long long)BATCH * SEQ * EMB) return;
  int c = (int)(i % EMB);
  long long bs = i / EMB;
  int tok = ids[bs];
  e[i] = (__bf16)tab[(size_t)tok * EMB + c];
}

// Time-major zero-padded copy of final h for conv0's free im2col.
__global__ void build_gpad_kernel(const __bf16* __restrict__ Hb, __bf16* __restrict__ g) {
  long long i = blockIdx.x * (long long)blockDim.x + threadIdx.x;
  if (i >= (long long)BATCH * 207 * KP) return;
  int c = (int)(i % KP);
  long long bt = i / KP;
  int tt = (int)(bt % 207);
  int b  = (int)(bt / 207);
  __bf16 v = (__bf16)0.0f;
  if (tt >= 1 && tt <= NPER) v = Hb[((size_t)(b * NPER + tt - 1)) * KP + c];
  g[i] = v;
}

// --------------------------- GGCN: m = h @ W  (2x2 blocked) ----------------
__global__ void gemm_hW_kernel(const __bf16* __restrict__ A,   // [NNODE, KP] bf16
                               const __bf16* __restrict__ Bp,  // packed [7][NTP]
                               float* __restrict__ C) {        // [NNODE, HP] f32
  const int MT2 = NNODE / 32;                 // 820 row-tile pairs
  const int NT2 = NTP / 2;                    // 7 col-tile pairs
  const int tid = threadIdx.x, lane = tid & 31, wave = tid >> 5;
  int tile = blockIdx.x * 8 + wave;
  if (tile >= MT2 * NT2) return;
  int mt2 = tile / NT2, nt2 = tile % NT2;
  int nt  = nt2 * 2;
  int row0 = mt2 * 32 + (lane & 15);
  int row1 = row0 + 16;
  v8f a00 = {}, a01 = {}, a10 = {}, a11 = {};
#pragma unroll
  for (int kt = 0; kt < KP / 32; ++kt) {
    v16bf fa0 = load_a_frag(A, row0, KP, kt * 32, lane);
    v16bf fa1 = load_a_frag(A, row1, KP, kt * 32, lane);
    v16bf fb0 = load_b_frag(Bp, kt, nt,     NTP, lane);
    v16bf fb1 = load_b_frag(Bp, kt, nt + 1, NTP, lane);
    a00 = wmma_bf16(fa0, fb0, a00);
    a01 = wmma_bf16(fa0, fb1, a01);
    a10 = wmma_bf16(fa1, fb0, a10);
    a11 = wmma_bf16(fa1, fb1, a11);
  }
  int hi = lane >> 4;
#pragma unroll
  for (int j2 = 0; j2 < 2; ++j2) {
    int col = (nt + j2) * 16 + (lane & 15);
    if (col >= HP) continue;                  // tile 13 is pure padding
#pragma unroll
    for (int j = 0; j < 8; ++j) {
      int r = mt2 * 32 + hi * 8 + j;
      C[(size_t)r * HP + col]        = j2 ? a01[j] : a00[j];
      C[(size_t)(r + 16) * HP + col] = j2 ? a11[j] : a10[j];
    }
  }
}

// --------------------------- edge scatter-add ------------------------------
__global__ void scatter_add_kernel(const int* __restrict__ ei,
                                   const float* __restrict__ m,
                                   float* __restrict__ agg) {
  long long i = blockIdx.x * (long long)blockDim.x + threadIdx.x;
  if (i >= (long long)NEDGE * HDIM) return;
  int e = (int)(i / HDIM), c = (int)(i % HDIM);
  int src = ei[e], dst = ei[NEDGE + e];
  atomicAdd(&agg[(size_t)dst * HP + c], m[(size_t)src * HP + c]);  // f32 atomic, hits L2
}

// --------------------------- fused GRU cell --------------------------------
// One block = one 16-row tile. 8 waves cooperatively compute gi=agg@Wih^T and
// gh=h@Whh^T over 38 column tiles (processed as 19 pairs; each A fragment pair
// feeds 4 WMMAs) into dynamic LDS, then the block applies the GRU nonlinearity
// and writes the new h (f32 + bf16, in place).
__global__ void gru_fused_kernel(__bf16* __restrict__ Hb,          // in/out [NNODE,KP]
                                 const __bf16* __restrict__ Aggb,  // [NNODE,KP]
                                 const __bf16* __restrict__ Bih,   // packed [7][38]
                                 const __bf16* __restrict__ Bhh,   // packed [7][38]
                                 const float* __restrict__ bih,
                                 const float* __restrict__ bhh,
                                 float* __restrict__ Hf) {         // in/out [NNODE,HP]
  extern __shared__ float sm[];
  float* sgi = sm;                // [16][608]
  float* sgh = sm + 16 * G3;      // [16][608]
  const int tid = threadIdx.x, lane = tid & 31, wave = tid >> 5;
  const int mrow = blockIdx.x * 16;
  const int arow = mrow + (lane & 15);

  for (int cp = wave; cp < (G3 / 16) / 2; cp += 8) {   // 19 column-tile pairs
    int ct0 = cp * 2, ct1 = cp * 2 + 1;
    v8f gi0 = {}, gi1 = {}, gh0 = {}, gh1 = {};
#pragma unroll
    for (int kt = 0; kt < KP / 32; ++kt) {
      v16bf ah = load_a_frag(Hb,   arow, KP, kt * 32, lane);
      v16bf aa = load_a_frag(Aggb, arow, KP, kt * 32, lane);
      v16bf bi0 = load_b_frag(Bih, kt, ct0, G3 / 16, lane);
      v16bf bi1 = load_b_frag(Bih, kt, ct1, G3 / 16, lane);
      v16bf bh0 = load_b_frag(Bhh, kt, ct0, G3 / 16, lane);
      v16bf bh1 = load_b_frag(Bhh, kt, ct1, G3 / 16, lane);
      gi0 = wmma_bf16(aa, bi0, gi0);   // agg @ wih^T
      gi1 = wmma_bf16(aa, bi1, gi1);
      gh0 = wmma_bf16(ah, bh0, gh0);   // h @ whh^T
      gh1 = wmma_bf16(ah, bh1, gh1);
    }
    int hi = lane >> 4;
    int col0 = ct0 * 16 + (lane & 15), col1 = ct1 * 16 + (lane & 15);
#pragma unroll
    for (int j = 0; j < 8; ++j) {
      int r = hi * 8 + j;
      sgi[r * G3 + col0] = gi0[j];
      sgi[r * G3 + col1] = gi1[j];
      sgh[r * G3 + col0] = gh0[j];
      sgh[r * G3 + col1] = gh1[j];
    }
  }
  __syncthreads();

  for (int idx = tid; idx < 16 * HDIM; idx += 256) {
    int r = idx / HDIM, c = idx % HDIM;
    float ir = sgi[r * G3 + c]            + bih[c];
    float iz = sgi[r * G3 + HDIM + c]     + bih[HDIM + c];
    float in = sgi[r * G3 + 2 * HDIM + c] + bih[2 * HDIM + c];
    float hr = sgh[r * G3 + c]            + bhh[c];
    float hz = sgh[r * G3 + HDIM + c]     + bhh[HDIM + c];
    float hn = sgh[r * G3 + 2 * HDIM + c] + bhh[2 * HDIM + c];
    float rg = 1.0f / (1.0f + __expf(-(ir + hr)));
    float zg = 1.0f / (1.0f + __expf(-(iz + hz)));
    float ng = tanhf(in + rg * hn);
    size_t gr = (size_t)(mrow + r);
    float hold = Hf[gr * HP + c];
    float hnew = (1.0f - zg) * ng + zg * hold;
    Hf[gr * HP + c] = hnew;
    Hb[gr * KP + c] = (__bf16)hnew;
  }
}

// --------------------------- fused conv1d + relu + max ---------------------
// Free-im2col GEMM, 2x2 blocked: output row t reads Kr contiguous bf16 at
// A + t*lda. Each wave owns a 32x32 block of the output; after the K loop it
// bias+relu's, reduces max over valid rows (shfl across the half-wave), and
// atomically max-merges into feats[b][foff + col].
__global__ void conv_fused_kernel(const __bf16* __restrict__ A0, long long abstride,
                                  int lda, const __bf16* __restrict__ Bp, int KT,
                                  int Tvalid, const float* __restrict__ bias,
                                  float* __restrict__ feats, int foff) {
  const int NT2 = NTP / 2;   // 7 col-tile pairs
  int b = blockIdx.y;
  const __bf16* A = A0 + (long long)b * abstride;
  const int tid = threadIdx.x, lane = tid & 31, wave = tid >> 5;
  int MT2 = (Tvalid + 31) / 32;
  int tile = blockIdx.x * 8 + wave;
  if (tile >= MT2 * NT2) return;
  int mt2 = tile / NT2, nt = (tile % NT2) * 2;
  int t0 = mt2 * 32 + (lane & 15);
  int t1 = t0 + 16;
  if (t0 > Tvalid - 1) t0 = Tvalid - 1;        // clamp; masked out of the max
  if (t1 > Tvalid - 1) t1 = Tvalid - 1;
  v8f a00 = {}, a01 = {}, a10 = {}, a11 = {};
  for (int kt = 0; kt < KT; ++kt) {
    if (kt + 1 < KT) {
      __builtin_prefetch(A + (size_t)t0 * lda + (kt + 1) * 32, 0, 0);
      __builtin_prefetch(A + (size_t)t1 * lda + (kt + 1) * 32, 0, 0);
    }
    v16bf fa0 = load_a_frag(A, t0, lda, kt * 32, lane);
    v16bf fa1 = load_a_frag(A, t1, lda, kt * 32, lane);
    v16bf fb0 = load_b_frag(Bp, kt, nt,     NTP, lane);
    v16bf fb1 = load_b_frag(Bp, kt, nt + 1, NTP, lane);
    a00 = wmma_bf16(fa0, fb0, a00);
    a01 = wmma_bf16(fa0, fb1, a01);
    a10 = wmma_bf16(fa1, fb0, a10);
    a11 = wmma_bf16(fa1, fb1, a11);
  }
  int hi = lane >> 4;
#pragma unroll
  for (int j2 = 0; j2 < 2; ++j2) {
    float vmax = -1e30f;
#pragma unroll
    for (int j = 0; j < 8; ++j) {
      int m0 = mt2 * 32 + hi * 8 + j;
      if (m0 < Tvalid)      vmax = fmaxf(vmax, j2 ? a01[j] : a00[j]);
      if (m0 + 16 < Tvalid) vmax = fmaxf(vmax, j2 ? a11[j] : a10[j]);
    }
    vmax = fmaxf(vmax, __shfl_xor(vmax, 16, 32));  // merge the two row halves
    int colN = (nt + j2) * 16 + (lane & 15);
    if (lane < 16 && colN < HDIM) {
      float v = fmaxf(vmax + bias[colN], 0.0f);    // relu; monotone so max commutes
      atomicMax((int*)&feats[(size_t)b * 800 + foff + colN], __float_as_int(v));
    }
  }
}

// --------------------------- fused FC head ---------------------------------
__global__ void fc_fused_kernel(const float* __restrict__ feats,
                                const float* __restrict__ w1, const float* __restrict__ b1,
                                const float* __restrict__ w2, const float* __restrict__ b2,
                                const float* __restrict__ w3, const float* __restrict__ b3,
                                float* __restrict__ out) {
  __shared__ float f[800];
  __shared__ float h1[256];
  __shared__ float h2[128];
  int b = blockIdx.x, tid = threadIdx.x;
  for (int i = tid; i < 800; i += 256) f[i] = feats[(size_t)b * 800 + i];
  __syncthreads();
  { float s = b1[tid];
    for (int i = 0; i < 800; ++i) s += w1[(size_t)tid * 800 + i] * f[i];
    h1[tid] = fmaxf(s, 0.0f); }
  __syncthreads();
  if (tid < 128) {
    float s = b2[tid];
    for (int i = 0; i < 256; ++i) s += w2[(size_t)tid * 256 + i] * h1[i];
    h2[tid] = fmaxf(s, 0.0f);
  }
  __syncthreads();
  if (tid < 2) {
    float s = b3[tid];
    for (int i = 0; i < 128; ++i) s += w3[(size_t)tid * 128 + i] * h2[i];
    out[(size_t)b * 2 + tid] = s;
  }
}

// ---------------------------------------------------------------------------
extern "C" void kernel_launch(void* const* d_in, const int* in_sizes, int n_in,
                              void* d_out, int out_size, void* d_ws, size_t ws_size,
                              hipStream_t stream) {
  (void)in_sizes; (void)n_in; (void)out_size; (void)ws_size;
  const float* x    = (const float*)d_in[0];
  const int*   ei   = (const int*)  d_in[1];
  const int*   ids  = (const int*)  d_in[2];
  const float* tab  = (const float*)d_in[3];
  const float* ggc  = (const float*)d_in[4];
  const float* wih  = (const float*)d_in[5];
  const float* whh  = (const float*)d_in[6];
  const float* bih  = (const float*)d_in[7];
  const float* bhh  = (const float*)d_in[8];
  const float* c0w  = (const float*)d_in[9];
  const float* c0b  = (const float*)d_in[10];
  const float* c1w  = (const float*)d_in[11];
  const float* c1b  = (const float*)d_in[12];
  const float* c2w  = (const float*)d_in[13];
  const float* c2b  = (const float*)d_in[14];
  const float* c3w  = (const float*)d_in[15];
  const float* c3b  = (const float*)d_in[16];
  const float* f1w  = (const float*)d_in[17];
  const float* f1b  = (const float*)d_in[18];
  const float* f2w  = (const float*)d_in[19];
  const float* f2b  = (const float*)d_in[20];
  const float* f3w  = (const float*)d_in[21];
  const float* f3b  = (const float*)d_in[22];
  float* out = (float*)d_out;

  // ---- workspace carve-out (~208 MB) ----
  char* ws = (char*)d_ws;
  size_t off = 0;
  auto take = [&](size_t bytes) -> void* {
    void* p = ws + off;
    off = (off + bytes + 255) & ~(size_t)255;
    return p;
  };
  float*  Hf   = (float*) take((size_t)NNODE * HP * 4);
  __bf16* Hb   = (__bf16*)take((size_t)NNODE * KP * 2);
  float*  M    = (float*) take((size_t)NNODE * HP * 4);
  float*  Agg  = (float*) take((size_t)NNODE * HP * 4);
  __bf16* Aggb = (__bf16*)take((size_t)NNODE * KP * 2);
  __bf16* Eb   = (__bf16*)take((size_t)BATCH * SEQ * EMB * 2);
  __bf16* Gpad = (__bf16*)take((size_t)BATCH * 207 * KP * 2);
  float*  Feat = (float*) take((size_t)BATCH * 800 * 4);
  __bf16* Bggc = (__bf16*)take((size_t)NLAYER * 7 * NTP * 512 * 2);
  __bf16* Bih  = (__bf16*)take((size_t)7 * 38 * 512 * 2);
  __bf16* Bhh  = (__bf16*)take((size_t)7 * 38 * 512 * 2);
  __bf16* Bc0  = (__bf16*)take((size_t)21 * NTP * 512 * 2);
  __bf16* Bc1  = (__bf16*)take((size_t)72 * NTP * 512 * 2);
  __bf16* Bc2  = (__bf16*)take((size_t)96 * NTP * 512 * 2);
  __bf16* Bc3  = (__bf16*)take((size_t)120 * NTP * 512 * 2);

  auto cdiv = [](long long a, long long b) -> unsigned { return (unsigned)((a + b - 1) / b); };

  // ---- pack weights into WMMA fragment order (every call; deterministic) ----
  for (int l = 0; l < NLAYER; ++l)
    pack_b_kernel<<<cdiv(7LL * NTP * 512, 256), 256, 0, stream>>>(
        Bggc + (size_t)l * 7 * NTP * 512, ggc + (size_t)l * HDIM * HDIM, 7, NTP, 0, 0);
  pack_b_kernel<<<cdiv(7LL * 38 * 512, 256), 256, 0, stream>>>(Bih, wih, 7, 38, 1, 0);
  pack_b_kernel<<<cdiv(7LL * 38 * 512, 256), 256, 0, stream>>>(Bhh, whh, 7, 38, 1, 0);
  pack_b_kernel<<<cdiv(72LL * NTP * 512, 256), 256, 0, stream>>>(Bc1, c1w, 72, NTP, 2, 3);
  pack_b_kernel<<<cdiv(96LL * NTP * 512, 256), 256, 0, stream>>>(Bc2, c2w, 96, NTP, 2, 4);
  pack_b_kernel<<<cdiv(120LL * NTP * 512, 256), 256, 0, stream>>>(Bc3, c3w, 120, NTP, 2, 5);
  pack_b_kernel<<<cdiv(21LL * NTP * 512, 256), 256, 0, stream>>>(Bc0, c0w, 21, NTP, 3, 0);

  // ---- embedding gather + h init ----
  embed_gather_kernel<<<cdiv((long long)BATCH * SEQ * EMB, 256), 256, 0, stream>>>(ids, tab, Eb);
  init_h_kernel<<<cdiv((long long)NNODE * KP, 256), 256, 0, stream>>>(x, Hf, Hb);

  // ---- 6 GGCN layers ----
  const unsigned gemm_blocks = cdiv((long long)(NNODE / 32) * (NTP / 2), 8);
  for (int l = 0; l < NLAYER; ++l) {
    zero_f32_kernel<<<cdiv((long long)NNODE * HP, 256), 256, 0, stream>>>(Agg, (long long)NNODE * HP);
    gemm_hW_kernel<<<gemm_blocks, 256, 0, stream>>>(Hb, Bggc + (size_t)l * 7 * NTP * 512, M);
    scatter_add_kernel<<<cdiv((long long)NEDGE * HDIM, 256), 256, 0, stream>>>(ei, M, Agg);
    cvt_agg_kernel<<<cdiv((long long)NNODE * KP, 256), 256, 0, stream>>>(Agg, Aggb);
    gru_fused_kernel<<<NNODE / 16, 256, 2 * 16 * G3 * sizeof(float), stream>>>(
        Hb, Aggb, Bih, Bhh, bih, bhh, Hf);
  }

  // ---- text CNN + conv0, fused with bias/relu/max-pool ----
  build_gpad_kernel<<<cdiv((long long)BATCH * 207 * KP, 256), 256, 0, stream>>>(Hb, Gpad);
  zero_f32_kernel<<<cdiv((long long)BATCH * 800, 256), 256, 0, stream>>>(Feat, (long long)BATCH * 800);

  {
    dim3 g1(cdiv((long long)((510 + 31) / 32) * (NTP / 2), 8), BATCH);
    conv_fused_kernel<<<g1, 256, 0, stream>>>(Eb, (long long)SEQ * EMB, EMB, Bc1, 72, 510, c1b, Feat, 200);
    dim3 g2(cdiv((long long)((509 + 31) / 32) * (NTP / 2), 8), BATCH);
    conv_fused_kernel<<<g2, 256, 0, stream>>>(Eb, (long long)SEQ * EMB, EMB, Bc2, 96, 509, c2b, Feat, 400);
    dim3 g3(cdiv((long long)((508 + 31) / 32) * (NTP / 2), 8), BATCH);
    conv_fused_kernel<<<g3, 256, 0, stream>>>(Eb, (long long)SEQ * EMB, EMB, Bc3, 120, 508, c3b, Feat, 600);
    dim3 g0(cdiv((long long)((205 + 31) / 32) * (NTP / 2), 8), BATCH);
    conv_fused_kernel<<<g0, 256, 0, stream>>>(Gpad, (long long)207 * KP, KP, Bc0, 21, 205, c0b, Feat, 0);
  }

  // ---- FC head ----
  fc_fused_kernel<<<BATCH, 256, 0, stream>>>(Feat, f1w, f1b, f2w, f2b, f3w, f3b, out);
}
